// AHCRFuse_79568564126349
// MI455X (gfx1250) — matrix operand order repaired and checked by reading, not compiled, not run on hardware
//
#include <hip/hip_runtime.h>
#include <hip/hip_fp16.h>
#include <cmath>

typedef _Float16 h16 __attribute__((ext_vector_type(16)));
typedef _Float16 h8  __attribute__((ext_vector_type(8)));
typedef float    f8  __attribute__((ext_vector_type(8)));

#define NHEADS 8

__device__ __forceinline__ h16 cat8(h8 lo, h8 hi) {
  return __builtin_shufflevector(lo, hi, 0,1,2,3,4,5,6,7,8,9,10,11,12,13,14,15);
}

// ---------------- elementwise converts ----------------

__global__ void k_cvt_f16(const float* __restrict__ src, _Float16* __restrict__ dst, int n) {
  int i = blockIdx.x * 256 + threadIdx.x;
  if (i < n) dst[i] = (_Float16)src[i];
}

// [B,C,HW] f32 -> [B*HW, C] f16   (flatten + transpose)
__global__ void k_flatten_f16(const float* __restrict__ src, _Float16* __restrict__ dst,
                              int B, int C, int HW) {
  int i = blockIdx.x * 256 + threadIdx.x;
  int total = B * C * HW;
  if (i >= total) return;
  int c = i % C; int t = i / C; int n = t % HW; int b = t / HW;
  dst[i] = (_Float16)src[((size_t)b * C + c) * HW + n];
}

// ---------------- WMMA GEMM: C[M,N] = A[M,K] @ B[K,N], f16 in ----------------
// Block tile 128x64, 8 waves of 32x32 (4 WMMA per wave per K-step).
// MODE 0: f32 store  1: f32 accumulate  2: f16 store  3: f16 transposed store [N,M]
template <int MODE>
__global__ __launch_bounds__(256) void k_wmma_gemm(
    const _Float16* __restrict__ A, const _Float16* __restrict__ Bw,
    void* __restrict__ Cp, int M, int N, int K)
{
  __shared__ __align__(16) _Float16 As[128][40];
  __shared__ __align__(16) _Float16 Bs[64][40];   // transposed tile: [n][k]
  const int tid = threadIdx.x;
  const int bm = blockIdx.y * 128, bn = blockIdx.x * 64;
  const int w = tid >> 5, lane = tid & 31, lh = lane & 15, hi = lane >> 4;
  const int wm = (w >> 1) * 32, wn = (w & 1) * 32;
  f8 acc[2][2] = {};
  // A loader: 128x32 halves = 512 16B-chunks -> 2 per thread (async to LDS)
  const int ar0 = tid >> 2,         ac0 = (tid & 3) * 8;
  const int ar1 = (tid + 256) >> 2, ac1 = ((tid + 256) & 3) * 8;
  const int gr0 = bm + ar0, gr1 = bm + ar1;
  const int cr0 = gr0 < M ? gr0 : M - 1;           // clamp (garbage rows masked at store)
  const int cr1 = gr1 < M ? gr1 : M - 1;
  // B loader: 32x64 halves, 8 per thread, transposed into LDS
  const int bk = tid >> 3, bn8 = (tid & 7) * 8;
  const unsigned lds_a0 = (unsigned)(uintptr_t)&As[ar0][ac0];
  const unsigned lds_a1 = (unsigned)(uintptr_t)&As[ar1][ac1];

  for (int kk = 0; kk < K; kk += 32) {
    // stage A: async global -> LDS (ASYNCcnt), 16B per lane per op
    {
      const _Float16* ga0 = A + (size_t)cr0 * K + kk + ac0;
      const _Float16* ga1 = A + (size_t)cr1 * K + kk + ac1;
      asm volatile("global_load_async_to_lds_b128 %0, %1, off"
                   :: "v"(lds_a0), "v"(ga0) : "memory");
      asm volatile("global_load_async_to_lds_b128 %0, %1, off"
                   :: "v"(lds_a1), "v"(ga1) : "memory");
    }
    // stage B 32x64, transposed into LDS
    h8 bv = *(const h8*)(Bw + (size_t)(kk + bk) * N + bn + bn8);
    #pragma unroll
    for (int i = 0; i < 8; ++i) Bs[bn8 + i][bk] = bv[i];
    if (kk + 32 < K) {
      __builtin_prefetch(A + (size_t)cr0 * K + kk + 32 + ac0, 0, 3);
      __builtin_prefetch(Bw + (size_t)(kk + 32 + bk) * N + bn + bn8, 0, 3);
    }
    asm volatile("s_wait_asynccnt 0" ::: "memory");
    __syncthreads();

    // A fragments (16x32): lane<16 -> K 0..7 & 16..23 ; lane>=16 -> K 8..15 & 24..31
    h16 af[2], bf[2];
    #pragma unroll
    for (int i = 0; i < 2; ++i) {
      af[i] = cat8(*(const h8*)&As[wm + i * 16 + lh][hi ? 8 : 0],
                   *(const h8*)&As[wm + i * 16 + lh][hi ? 24 : 16]);
      bf[i] = cat8(*(const h8*)&Bs[wn + i * 16 + lh][hi * 8],
                   *(const h8*)&Bs[wn + i * 16 + lh][16 + hi * 8]);
    }
    #pragma unroll
    for (int i = 0; i < 2; ++i)
      #pragma unroll
      for (int j = 0; j < 2; ++j)
        acc[i][j] = __builtin_amdgcn_wmma_f32_16x16x32_f16(
            false, af[i], false, bf[j], (short)0, acc[i][j], false, false);
    __syncthreads();
  }

  #pragma unroll
  for (int i = 0; i < 2; ++i) {
    const int mrow = bm + wm + i * 16 + hi * 8;   // rows mrow..mrow+7 (M % 16 == 0)
    if (mrow >= M) continue;
    #pragma unroll
    for (int j = 0; j < 2; ++j) {
      const int nc = bn + wn + j * 16 + lh;
      if (MODE == 0 || MODE == 1) {
        float* C32 = (float*)Cp;
        #pragma unroll
        for (int r = 0; r < 8; ++r) {
          size_t ix = (size_t)(mrow + r) * N + nc;
          if (MODE == 1) C32[ix] += acc[i][j][r];
          else           C32[ix]  = acc[i][j][r];
        }
      } else if (MODE == 2) {
        _Float16* C16 = (_Float16*)Cp;
        #pragma unroll
        for (int r = 0; r < 8; ++r)
          C16[(size_t)(mrow + r) * N + nc] = (_Float16)acc[i][j][r];
      } else {
        _Float16* Ct = (_Float16*)Cp;   // [N][M] row-major
        h8 v;
        #pragma unroll
        for (int r = 0; r < 8; ++r) v[r] = (_Float16)acc[i][j][r];
        *(h8*)(Ct + (size_t)nc * M + mrow) = v;
      }
    }
  }
}

// ---------------- flash attention (64 keys / iteration) ----------------
// Q,K: [B*N, C] f16 row-major; Vt: [C, B*N] f16; O: [B*N, C] f16
template <int HD>
__global__ __launch_bounds__(128) void k_flash(
    const _Float16* __restrict__ Qp, const _Float16* __restrict__ Kp,
    const _Float16* __restrict__ Vt, _Float16* __restrict__ Op,
    int Ntok, float scale)
{
  constexpr int C = HD * NHEADS;
  constexpr int NQF = HD / 32;
  constexpr int NOD = HD / 16;
  __shared__ __align__(16) _Float16 Plds[4][16][72];   // per-wave 16x64 P tile
  const int w = threadIdx.x >> 5, lane = threadIdx.x & 31, lh = lane & 15, hi = lane >> 4;
  const int qBase = blockIdx.x * 64 + w * 16;
  if (qBase >= Ntok) return;
  const int h = blockIdx.y, b = blockIdx.z;
  const int BN = (int)gridDim.z * Ntok;
  const int bNtok = b * Ntok;
  const int hHD = h * HD;

  h16 qf[NQF];
  {
    size_t rq = (size_t)(bNtok + qBase + lh) * C + hHD;
    #pragma unroll
    for (int f = 0; f < NQF; ++f)
      qf[f] = cat8(*(const h8*)(Qp + rq + f * 32 + hi * 8),
                   *(const h8*)(Qp + rq + f * 32 + 16 + hi * 8));
  }
  f8 zero = {};
  f8 acc[NOD];
  #pragma unroll
  for (int d = 0; d < NOD; ++d) acc[d] = zero;
  float mrun[8], lrun[8];
  #pragma unroll
  for (int r = 0; r < 8; ++r) { mrun[r] = -1e30f; lrun[r] = 0.f; }

  const int nkt = (Ntok + 63) / 64;
  for (int kt = 0; kt < nkt; ++kt) {
    const int ktB = kt * 64;
    f8 s[4]; bool ok[4]; size_t rk[4];
    #pragma unroll
    for (int t = 0; t < 4; ++t) {
      s[t] = zero;
      int key = ktB + t * 16 + lh;
      ok[t] = key < Ntok;
      rk[t] = (size_t)(bNtok + (ok[t] ? key : Ntok - 1)) * C + hHD;
    }
    #pragma unroll
    for (int f = 0; f < NQF; ++f) {
      const int k0 = f * 32 + hi * 8;   // B fragment: K k0..k0+7 & k0+16..k0+23 of a K-row
      #pragma unroll
      for (int t = 0; t < 4; ++t) {
        h16 bfr = cat8(*(const h8*)(Kp + rk[t] + k0), *(const h8*)(Kp + rk[t] + k0 + 16));
        s[t] = __builtin_amdgcn_wmma_f32_16x16x32_f16(false, qf[f], false, bfr,
                                                      (short)0, s[t], false, false);
      }
    }
    float corr[8];
    #pragma unroll
    for (int r = 0; r < 8; ++r) {
      float v[4];
      #pragma unroll
      for (int t = 0; t < 4; ++t) v[t] = ok[t] ? s[t][r] * scale : -1e30f;
      float tm = fmaxf(fmaxf(v[0], v[1]), fmaxf(v[2], v[3]));
      tm = fmaxf(tm, __shfl_xor(tm, 1, 32));
      tm = fmaxf(tm, __shfl_xor(tm, 2, 32));
      tm = fmaxf(tm, __shfl_xor(tm, 4, 32));
      tm = fmaxf(tm, __shfl_xor(tm, 8, 32));
      float nm = fmaxf(mrun[r], tm);
      float c  = __expf(mrun[r] - nm);
      float rs = 0.f;
      #pragma unroll
      for (int t = 0; t < 4; ++t) {
        float p = __expf(v[t] - nm);
        rs += p;
        Plds[w][hi * 8 + r][t * 16 + lh] = (_Float16)p;   // C/D -> row-major P tile
      }
      rs += __shfl_xor(rs, 1, 32);
      rs += __shfl_xor(rs, 2, 32);
      rs += __shfl_xor(rs, 4, 32);
      rs += __shfl_xor(rs, 8, 32);
      lrun[r] = lrun[r] * c + rs;
      mrun[r] = nm;
      corr[r] = c;
    }
    #pragma unroll
    for (int d = 0; d < NOD; ++d)
      #pragma unroll
      for (int r = 0; r < 8; ++r) acc[d][r] *= corr[r];
    asm volatile("s_wait_dscnt 0" ::: "memory");   // cross-lane LDS RAW within wave
    h16 pf0 = cat8(*(const h8*)&Plds[w][lh][hi * 8],
                   *(const h8*)&Plds[w][lh][16 + hi * 8]);
    h16 pf1 = cat8(*(const h8*)&Plds[w][lh][32 + hi * 8],
                   *(const h8*)&Plds[w][lh][48 + hi * 8]);
    #pragma unroll
    for (int d = 0; d < NOD; ++d) {
      size_t vb = (size_t)(hHD + d * 16 + lh) * BN + bNtok + ktB + hi * 8;
      h16 vf0 = cat8(*(const h8*)(Vt + vb),      *(const h8*)(Vt + vb + 16));
      h16 vf1 = cat8(*(const h8*)(Vt + vb + 32), *(const h8*)(Vt + vb + 48));
      acc[d] = __builtin_amdgcn_wmma_f32_16x16x32_f16(false, pf0, false, vf0,
                                                      (short)0, acc[d], false, false);
      acc[d] = __builtin_amdgcn_wmma_f32_16x16x32_f16(false, pf1, false, vf1,
                                                      (short)0, acc[d], false, false);
    }
  }
  #pragma unroll
  for (int r = 0; r < 8; ++r) {
    const float inv = 1.f / lrun[r];
    const int m = qBase + hi * 8 + r;
    #pragma unroll
    for (int d = 0; d < NOD; ++d)
      Op[(size_t)(bNtok + m) * C + hHD + d * 16 + lh] = (_Float16)(acc[d][r] * inv);
  }
}

// ---------------- epilogues ----------------

__global__ void k_proj_ep(const _Float16* __restrict__ Xq, const float* __restrict__ Acc,
                          const float* __restrict__ bias, const float* __restrict__ gamma,
                          _Float16* __restrict__ Y, int total, int C) {
  int i = blockIdx.x * 256 + threadIdx.x;
  if (i >= total) return;
  float o = Acc[i] + bias[i % C];
  Y[i] = (_Float16)((float)Xq[i] + gamma[0] * o);
}

// Acc [B*N, C] f32 -> Out [B, C, N] f32 with bias + SiLU
__global__ void k_fuse_ep(const float* __restrict__ Acc, const float* __restrict__ bias,
                          float* __restrict__ Out, int B, int C, int Ntok) {
  int i = blockIdx.x * 256 + threadIdx.x;
  int total = B * C * Ntok;
  if (i >= total) return;
  int n = i % Ntok; int t = i / Ntok; int o = t % C; int b = t / C;
  float y = Acc[((size_t)b * Ntok + n) * C + o] + bias[o];
  Out[i] = y / (1.f + __expf(-y));
}

// ---------------- host ----------------

static inline int cdiv(int a, int b) { return (a + b - 1) / b; }

struct AttnP { const float *wq, *wk, *wv, *pw, *pb, *g; };
struct ConvP { const float *w, *b; };

extern "C" void kernel_launch(void* const* d_in, const int* in_sizes, int n_in,
                              void* d_out, int out_size, void* d_ws, size_t ws_size,
                              hipStream_t stream) {
  const float* X[6];
  for (int i = 0; i < 6; ++i) X[i] = (const float*)d_in[i];

  AttnP ap[6]; ConvP cp[3];
  int base = 6;
  for (int a = 0; a < 6; ++a) {
    if (in_sizes[base] == 1) {           // alphabetical leaves: gamma, proj_b, proj_w, wk, wq, wv
      ap[a].g  = (const float*)d_in[base + 0];
      ap[a].pb = (const float*)d_in[base + 1];
      ap[a].pw = (const float*)d_in[base + 2];
      ap[a].wk = (const float*)d_in[base + 3];
      ap[a].wq = (const float*)d_in[base + 4];
      ap[a].wv = (const float*)d_in[base + 5];
    } else {                              // insertion order: wq, wk, wv, proj_w, proj_b, gamma
      ap[a].wq = (const float*)d_in[base + 0];
      ap[a].wk = (const float*)d_in[base + 1];
      ap[a].wv = (const float*)d_in[base + 2];
      ap[a].pw = (const float*)d_in[base + 3];
      ap[a].pb = (const float*)d_in[base + 4];
      ap[a].g  = (const float*)d_in[base + 5];
    }
    base += 6;
  }
  for (int f = 0; f < 3; ++f) {
    if (in_sizes[base] < in_sizes[base + 1]) { cp[f].b = (const float*)d_in[base];     cp[f].w = (const float*)d_in[base + 1]; }
    else                                     { cp[f].w = (const float*)d_in[base];     cp[f].b = (const float*)d_in[base + 1]; }
    base += 2;
  }

  // workspace carve-up (max BN*C = 4608*256 = 1,179,648 elements, level c3)
  const size_t MAXE = 4608ull * 256ull;
  char* ws = (char*)d_ws;
  size_t off = 0;
  auto alloc = [&](size_t bytes) -> void* {
    off = (off + 255) & ~(size_t)255;
    void* p = ws + off; off += bytes; return p;
  };
  _Float16* XA  = (_Float16*)alloc(MAXE * 2);
  _Float16* XB  = (_Float16*)alloc(MAXE * 2);
  _Float16* Qb  = (_Float16*)alloc(MAXE * 2);
  _Float16* Kb  = (_Float16*)alloc(MAXE * 2);
  _Float16* Vt  = (_Float16*)alloc(MAXE * 2 + 4096);  // pad: masked-key overread
  _Float16* Ob  = (_Float16*)alloc(MAXE * 2);
  _Float16* Yar = (_Float16*)alloc(MAXE * 2);
  _Float16* Ybr = (_Float16*)alloc(MAXE * 2);
  float*    Acc = (float*)   alloc(MAXE * 4);
  _Float16* WQ  = (_Float16*)alloc(1024ull * 1024 * 2);
  _Float16* WK  = (_Float16*)alloc(1024ull * 1024 * 2);
  _Float16* WV  = (_Float16*)alloc(1024ull * 1024 * 2);
  _Float16* WP  = (_Float16*)alloc(1024ull * 1024 * 2);
  _Float16* WF  = (_Float16*)alloc(2048ull * 1024 * 2);
  (void)ws_size; (void)n_in; (void)out_size;

  const int Cs[3] = {256, 512, 1024};
  const int Ns[3] = {2304, 576, 144};
  const int B = 2;
  float* outp = (float*)d_out;

  for (int lv = 0; lv < 3; ++lv) {
    const int C = Cs[lv], Nt = Ns[lv], BN = B * Nt, hd = C / NHEADS;
    const float scale = 1.0f / sqrtf((float)hd);
    const int tot = BN * C;

    k_flatten_f16<<<cdiv(tot, 256), 256, 0, stream>>>(X[lv * 2],     XA, B, C, Nt);
    k_flatten_f16<<<cdiv(tot, 256), 256, 0, stream>>>(X[lv * 2 + 1], XB, B, C, Nt);

    for (int dir = 0; dir < 2; ++dir) {
      const AttnP& p = ap[lv * 2 + dir];
      const _Float16* Xq  = dir ? XB : XA;
      const _Float16* Xkv = dir ? XA : XB;
      k_cvt_f16<<<cdiv(C * C, 256), 256, 0, stream>>>(p.wq, WQ, C * C);
      k_cvt_f16<<<cdiv(C * C, 256), 256, 0, stream>>>(p.wk, WK, C * C);
      k_cvt_f16<<<cdiv(C * C, 256), 256, 0, stream>>>(p.wv, WV, C * C);
      k_cvt_f16<<<cdiv(C * C, 256), 256, 0, stream>>>(p.pw, WP, C * C);

      dim3 gg(C / 64, cdiv(BN, 128));
      k_wmma_gemm<2><<<gg, 256, 0, stream>>>(Xq,  WQ, Qb, BN, C, C);
      k_wmma_gemm<2><<<gg, 256, 0, stream>>>(Xkv, WK, Kb, BN, C, C);
      k_wmma_gemm<3><<<gg, 256, 0, stream>>>(Xkv, WV, Vt, BN, C, C);   // [C, BN]

      dim3 fg(cdiv(Nt, 64), NHEADS, B);
      if (hd == 32)       k_flash<32> <<<fg, 128, 0, stream>>>(Qb, Kb, Vt, Ob, Nt, scale);
      else if (hd == 64)  k_flash<64> <<<fg, 128, 0, stream>>>(Qb, Kb, Vt, Ob, Nt, scale);
      else                k_flash<128><<<fg, 128, 0, stream>>>(Qb, Kb, Vt, Ob, Nt, scale);

      k_wmma_gemm<0><<<gg, 256, 0, stream>>>(Ob, WP, Acc, BN, C, C);
      k_proj_ep<<<cdiv(tot, 256), 256, 0, stream>>>(Xq, Acc, p.pb, p.g,
                                                    dir ? Ybr : Yar, tot, C);
    }

    k_cvt_f16<<<cdiv(2 * C * C, 256), 256, 0, stream>>>(cp[lv].w, WF, 2 * C * C);
    dim3 gg(C / 64, cdiv(BN, 128));
    k_wmma_gemm<0><<<gg, 256, 0, stream>>>(Yar, WF,                  Acc, BN, C, C);
    k_wmma_gemm<1><<<gg, 256, 0, stream>>>(Ybr, WF + (size_t)C * C,  Acc, BN, C, C);
    k_fuse_ep<<<cdiv(tot, 256), 256, 0, stream>>>(Acc, cp[lv].b, outp, B, C, Nt);
    outp += (size_t)tot;
  }
}
